// MultiHeadAttention_32040456028894
// MI455X (gfx1250) — matrix operand order repaired and checked by reading, not compiled, not run on hardware
//
#include <hip/hip_runtime.h>
#include <hip/hip_bf16.h>

// MultiHeadAttention (source bug preserved: Wq used for Q,K,V) on gfx1250.
// B=4, S=2048, D=512, H=8, HD=64. All GEMMs via v_wmma_f32_16x16x32_f16.
// K/V tiles streamed into LDS with the Tensor Data Mover, double-buffered so
// DMA of tile kt+1 overlaps compute of tile kt (s_wait_tensorcnt pipelining).

#define B_  4
#define S_  2048
#define D_  512
#define H_  8
#define HD_ 64

typedef __attribute__((ext_vector_type(16))) _Float16     v16h;
typedef __attribute__((ext_vector_type(8)))  _Float16     v8h;
typedef __attribute__((ext_vector_type(8)))  float        v8f;
typedef __attribute__((ext_vector_type(4)))  unsigned int v4u;
typedef __attribute__((ext_vector_type(8)))  int          v8i_t;
typedef __attribute__((ext_vector_type(4)))  int          v4i_t;

#if __has_builtin(__builtin_amdgcn_tensor_load_to_lds)
#define HAVE_TDM 1
#else
#define HAVE_TDM 0
#endif

union HF16 { v16h v; v8h h[2]; };

// ---- WMMA fragment loaders (ISA 7.12.2 layouts, wave32) ----
// A (16x32 f16, row-major in LDS): lane L<16 holds row M=L, K=0..7 & 16..23;
// lane L+16 holds row M=L, K=8..15 & 24..31.  => two 16B LDS loads per lane.
__device__ inline v16h frag_a_rm(const _Float16* base, int stride, int lane) {
  const int m  = lane & 15;
  const int hi = (lane >> 4) & 1;
  const _Float16* p = base + m * stride + hi * 8;
  HF16 u;
  u.h[0] = *(const v8h*)(p);
  u.h[1] = *(const v8h*)(p + 16);
  return u.v;
}
// B (32x16 f16) taken from column-major storage Bt[n][k] (stride in k-elems):
// lane n<16 holds col N=n, K=0..7 & 16..23; lane n+16 holds col N=n, K=8..15 & 24..31.
__device__ inline v16h frag_b_cm(const _Float16* bt, int stride, int lane) {
  const int n  = lane & 15;
  const int hi = (lane >> 4) & 1;
  const _Float16* p = bt + n * stride + hi * 8;
  HF16 u;
  u.h[0] = *(const v8h*)(p);
  u.h[1] = *(const v8h*)(p + 16);
  return u.v;
}

__device__ inline unsigned lds_off(const void* p) {
  // Flat pointers to LDS carry the LDS byte offset in the low 32 bits
  // (ISA 10.2).
  return (unsigned)(unsigned long long)p;
}

__device__ inline v8h pack8(const float4& x0, const float4& x1) {
  v8h h;
  h[0] = (_Float16)x0.x; h[1] = (_Float16)x0.y;
  h[2] = (_Float16)x0.z; h[3] = (_Float16)x0.w;
  h[4] = (_Float16)x1.x; h[5] = (_Float16)x1.y;
  h[6] = (_Float16)x1.z; h[7] = (_Float16)x1.w;
  return h;
}

// ---- Tensor Data Mover: 2D tile load, 16-bit elements (D# per ISA ch.8) ----
__device__ inline void tdm_load_2d(unsigned lds_byte, const void* g,
                                   unsigned tensor_d0, unsigned tensor_d1,
                                   unsigned tile_d0, unsigned tile_d1,
                                   unsigned row_stride_elems) {
#if HAVE_TDM
  unsigned long long ga = (unsigned long long)g;
  v4u g0;
  g0[0] = 1u;                                   // count=1, user mode, no gather
  g0[1] = lds_byte;                             // lds_addr
  g0[2] = (unsigned)ga;                         // global_addr[31:0]
  g0[3] = (unsigned)((ga >> 32) & 0x1FFFFFFu)   // global_addr[56:32]
          | (2u << 30);                         // type = 2 ("image")
  v8i_t g1;
  g1[0] = (int)(1u << 16);                      // wg_mask=0, data_size=1 (2B)
  g1[1] = (int)((tensor_d0 & 0xFFFFu) << 16);   // tensor_dim0 lo16 @bit48
  g1[2] = (int)(((tensor_d0 >> 16) & 0xFFFFu) | ((tensor_d1 & 0xFFFFu) << 16));
  g1[3] = (int)(((tensor_d1 >> 16) & 0xFFFFu) | ((tile_d0 & 0xFFFFu) << 16));
  g1[4] = (int)(tile_d1 & 0xFFFFu);             // tile_dim1 | tile_dim2=0
  g1[5] = (int)row_stride_elems;                // tensor_dim0_stride[31:0]
  g1[6] = 0;
  g1[7] = 0;
  v4i_t z4 = {0, 0, 0, 0};
  v8i_t z8 = {0, 0, 0, 0, 0, 0, 0, 0};
  __builtin_amdgcn_tensor_load_to_lds(g0, g1, z4, z4, z8, 0);
#else
  (void)lds_byte; (void)g; (void)tensor_d0; (void)tensor_d1;
  (void)tile_d0; (void)tile_d1; (void)row_stride_elems;
#endif
}

__device__ inline void coop_copy(_Float16* dst, const _Float16* src,
                                 int rows, int cols, int sstride,
                                 int tid, int nthr) {
  const int chunks = rows * cols / 8;
  for (int i = tid; i < chunks; i += nthr) {
    const int r = (i * 8) / cols;
    const int c = (i * 8) % cols;
    *(v8h*)(dst + r * cols + c) = *(const v8h*)(src + (size_t)r * sstride + c);
  }
}

// ================= GEMM: Y[M,512] = X[M,512] @ W[512,512] + bias =============
// Block: 128 threads (4 waves); block tile 128x64; each wave computes 32x64
// (2 A-frags x 4 B-frags = 8 WMMAs per 32-wide k-slab).
// OUT_MODE 0: f16 head-major [B,H,S,64]  (Q / K)
// OUT_MODE 1: f16 head-major transposed [B,H,64,S]  (V)
// OUT_MODE 2: f32 plain [M,512]  (final output)
template <bool IN_F32, int OUT_MODE>
__global__ __launch_bounds__(128) void gemm512(const void* __restrict__ X,
                                               const float* __restrict__ W,
                                               const float* __restrict__ bias,
                                               _Float16* __restrict__ Yh,
                                               float* __restrict__ Yf) {
  const int tid  = threadIdx.x;
  const int lane = tid & 31;
  const int wave = tid >> 5;
  const int m0   = blockIdx.y * 128;
  const int n0   = blockIdx.x * 64;

  __shared__ __align__(16) _Float16 Xs[128 * 32];  // [m][k]
  __shared__ __align__(16) _Float16 Ws[64 * 32];   // [n][k] (W transposed)

  v8f acc[2][4] = {};

  for (int k0 = 0; k0 < D_; k0 += 32) {
    {  // stage X (128x32) -> f16, 128-bit global loads
      const int r0 = tid >> 2;
      const int cb = (tid & 3) * 8;
#pragma unroll
      for (int rr = 0; rr < 4; ++rr) {
        const int r = r0 + rr * 32;
        if constexpr (IN_F32) {
          const float* xp = (const float*)X + (size_t)(m0 + r) * D_ + k0 + cb;
          const float4 x0 = *(const float4*)(xp);
          const float4 x1 = *(const float4*)(xp + 4);
          *(v8h*)(Xs + r * 32 + cb) = pack8(x0, x1);
        } else {
          const _Float16* xp = (const _Float16*)X + (size_t)(m0 + r) * D_ + k0 + cb;
          *(v8h*)(Xs + r * 32 + cb) = *(const v8h*)xp;
        }
      }
    }
    {  // stage W transposed (32k x 64n -> Ws[n][k]), 128-bit global loads
      const int kk = tid >> 2;
      const int nb = (tid & 3) * 16;
      const float* wp = W + (size_t)(k0 + kk) * D_ + n0 + nb;
#pragma unroll
      for (int jj = 0; jj < 4; ++jj) {
        const float4 w4 = *(const float4*)(wp + jj * 4);
        Ws[(nb + jj * 4 + 0) * 32 + kk] = (_Float16)w4.x;
        Ws[(nb + jj * 4 + 1) * 32 + kk] = (_Float16)w4.y;
        Ws[(nb + jj * 4 + 2) * 32 + kk] = (_Float16)w4.z;
        Ws[(nb + jj * 4 + 3) * 32 + kk] = (_Float16)w4.w;
      }
    }
    __syncthreads();

    // hoist all fragments of this k-slab, then back-to-back WMMAs
    const v16h a0 = frag_a_rm(Xs + (wave * 32 + 0) * 32, 32, lane);
    const v16h a1 = frag_a_rm(Xs + (wave * 32 + 16) * 32, 32, lane);
    v16h b[4];
#pragma unroll
    for (int nt = 0; nt < 4; ++nt) b[nt] = frag_b_cm(Ws + nt * 16 * 32, 32, lane);
#pragma unroll
    for (int nt = 0; nt < 4; ++nt) {
      acc[0][nt] = __builtin_amdgcn_wmma_f32_16x16x32_f16(
          false, a0, false, b[nt], (short)0, acc[0][nt], false, false);
      acc[1][nt] = __builtin_amdgcn_wmma_f32_16x16x32_f16(
          false, a1, false, b[nt], (short)0, acc[1][nt], false, false);
    }
    __syncthreads();
  }

  // epilogue: bias add + layout-specific store (C layout: VGPR i -> row i / i+8)
  const int hi8 = (lane >> 4) * 8;
  const int nc  = lane & 15;
#pragma unroll
  for (int mt = 0; mt < 2; ++mt) {
#pragma unroll
    for (int nt = 0; nt < 4; ++nt) {
      const int nG = n0 + nt * 16 + nc;
      const float bv = bias[nG];
#pragma unroll
      for (int i = 0; i < 8; ++i) {
        const int m = m0 + wave * 32 + mt * 16 + i + hi8;
        const float val = acc[mt][nt][i] + bv;
        if constexpr (OUT_MODE == 0) {
          const int b0 = m / S_, s = m % S_, h = nG / HD_, hd = nG % HD_;
          Yh[(((size_t)b0 * H_ + h) * S_ + s) * HD_ + hd] = (_Float16)val;
        } else if constexpr (OUT_MODE == 1) {
          const int b0 = m / S_, s = m % S_, h = nG / HD_, hd = nG % HD_;
          Yh[(((size_t)b0 * H_ + h) * HD_ + hd) * S_ + s] = (_Float16)val;
        } else {
          Yf[(size_t)m * D_ + nG] = val;
        }
      }
    }
  }
}

// ===================== Flash attention (online softmax) ======================
// Grid: (S/64, B*H). Block 128 threads = 4 waves; each wave owns 16 q-rows.
// K/V double-buffered in LDS; TDM for tile kt+1 issued before computing kt.
__global__ __launch_bounds__(128) void flash_attn(const _Float16* __restrict__ Qh,
                                                  const _Float16* __restrict__ Kh,
                                                  const _Float16* __restrict__ Vt,
                                                  _Float16* __restrict__ Ob) {
  const int tid  = threadIdx.x;
  const int lane = tid & 31;
  const int wave = tid >> 5;
  const int bh   = blockIdx.y;
  const int bb   = bh / H_;
  const int hh   = bh % H_;
  const int q0   = blockIdx.x * 64;
  const int NT   = S_ / 64;

  __shared__ __align__(16) _Float16 Qs[64 * 64];      // [q][hd]
  __shared__ __align__(16) _Float16 Ks[2][64 * 64];   // [key][hd], ping-pong
  __shared__ __align__(16) _Float16 Vs[2][64 * 64];   // [hd][key], ping-pong
  __shared__ __align__(16) _Float16 Ps[4][16 * 64];   // per-wave P tile

  const _Float16* Qb = Qh + (size_t)bh * S_ * HD_;
  const _Float16* Kb = Kh + (size_t)bh * S_ * HD_;
  const _Float16* Vb = Vt + (size_t)bh * HD_ * S_;

  float mrow[8], lrow[8];
#pragma unroll
  for (int i = 0; i < 8; ++i) { mrow[i] = -1e30f; lrow[i] = 0.0f; }
  v8f oacc[4] = {};
  v16h qa0 = {}, qa1 = {};  // Q fragments: loop-invariant, loaded at kt==0

#if HAVE_TDM
  if (wave == 0) {  // prime the pipeline: Q + tile 0
    tdm_load_2d(lds_off(Qs), Qb + (size_t)q0 * HD_, HD_, S_, HD_, 64, HD_);
    tdm_load_2d(lds_off(Ks[0]), Kb, HD_, S_, HD_, 64, HD_);
    tdm_load_2d(lds_off(Vs[0]), Vb, S_, HD_, 64, 64, S_);
  }
#else
  coop_copy(Qs, Qb + (size_t)q0 * HD_, 64, 64, HD_, tid, 128);
#endif

  for (int kt = 0; kt < NT; ++kt) {
    const int p = kt & 1;
#if HAVE_TDM
    __syncthreads();  // all waves done reading buffer p^1 (previous tile)
    if (wave == 0) {
      if (kt + 1 < NT) {  // prefetch next tile into the other buffer
        const int kn = (kt + 1) * 64;
        tdm_load_2d(lds_off(Ks[p ^ 1]), Kb + (size_t)kn * HD_, HD_, S_, HD_, 64, HD_);
        tdm_load_2d(lds_off(Vs[p ^ 1]), Vb + kn, S_, HD_, 64, 64, S_);
        // in-order TENSORcnt: allow the 2 just-issued to remain outstanding;
        // everything older (Q + current tile) is then complete.
        __builtin_amdgcn_s_wait_tensorcnt(2);
      } else {
        __builtin_amdgcn_s_wait_tensorcnt(0);
      }
    }
    __syncthreads();
#else
    __syncthreads();
    coop_copy(Ks[p], Kb + (size_t)(kt * 64) * HD_, 64, 64, HD_, tid, 128);
    coop_copy(Vs[p], Vb + kt * 64, 64, 64, S_, tid, 128);
    __syncthreads();
#endif

    if (kt == 0) {
      qa0 = frag_a_rm(Qs + (wave * 16) * HD_ + 0, HD_, lane);
      qa1 = frag_a_rm(Qs + (wave * 16) * HD_ + 32, HD_, lane);
    }

    // S = Q @ K^T  (K row-major [key][hd] is exactly the "Bt" operand layout)
    v8f sc[4] = {};
#pragma unroll
    for (int ksb = 0; ksb < 2; ++ksb) {
      const v16h a = (ksb == 0) ? qa0 : qa1;
      v16h kb[4];
#pragma unroll
      for (int nt = 0; nt < 4; ++nt)
        kb[nt] = frag_b_cm(Ks[p] + (nt * 16) * HD_ + ksb * 32, HD_, lane);
#pragma unroll
      for (int nt = 0; nt < 4; ++nt)
        sc[nt] = __builtin_amdgcn_wmma_f32_16x16x32_f16(
            false, a, false, kb[nt], (short)0, sc[nt], false, false);
    }

    // scale by 1/sqrt(HD) + per-row max (rows split lanes 0-15 / 16-31)
    float rmax[8];
#pragma unroll
    for (int i = 0; i < 8; ++i) rmax[i] = -1e30f;
#pragma unroll
    for (int nt = 0; nt < 4; ++nt)
#pragma unroll
      for (int i = 0; i < 8; ++i) {
        const float s = sc[nt][i] * 0.125f;
        sc[nt][i] = s;
        rmax[i] = fmaxf(rmax[i], s);
      }
#pragma unroll
    for (int i = 0; i < 8; ++i)
#pragma unroll
      for (int off = 8; off >= 1; off >>= 1)
        rmax[i] = fmaxf(rmax[i], __shfl_xor(rmax[i], off, 32));

    float alpha[8];
#pragma unroll
    for (int i = 0; i < 8; ++i) {
      const float mn = fmaxf(mrow[i], rmax[i]);
      alpha[i] = __expf(mrow[i] - mn);
      mrow[i] = mn;
    }

    float rsum[8];
#pragma unroll
    for (int i = 0; i < 8; ++i) rsum[i] = 0.0f;
#pragma unroll
    for (int nt = 0; nt < 4; ++nt)
#pragma unroll
      for (int i = 0; i < 8; ++i) {
        const float pv = __expf(sc[nt][i] - mrow[i]);
        sc[nt][i] = pv;
        rsum[i] += pv;
      }
#pragma unroll
    for (int i = 0; i < 8; ++i) {
#pragma unroll
      for (int off = 8; off >= 1; off >>= 1)
        rsum[i] += __shfl_xor(rsum[i], off, 32);
      lrow[i] = lrow[i] * alpha[i] + rsum[i];
    }
#pragma unroll
    for (int ot = 0; ot < 4; ++ot)
#pragma unroll
      for (int i = 0; i < 8; ++i) oacc[ot][i] *= alpha[i];

    // P -> LDS (wave-private) to re-fragment as WMMA-A
    _Float16* Pw = &Ps[wave][0];
    const int hi8 = (lane >> 4) * 8;
    const int nc  = lane & 15;
#pragma unroll
    for (int nt = 0; nt < 4; ++nt)
#pragma unroll
      for (int i = 0; i < 8; ++i)
        Pw[(i + hi8) * 64 + nt * 16 + nc] = (_Float16)sc[nt][i];

    // O += P @ V  (Vs[hd][key] is the "Bt" operand layout)
#pragma unroll
    for (int ksb = 0; ksb < 2; ++ksb) {
      const v16h a = frag_a_rm(Pw + ksb * 32, 64, lane);
      v16h vb[4];
#pragma unroll
      for (int ot = 0; ot < 4; ++ot)
        vb[ot] = frag_b_cm(Vs[p] + (ot * 16) * 64 + ksb * 32, 64, lane);
#pragma unroll
      for (int ot = 0; ot < 4; ++ot)
        oacc[ot] = __builtin_amdgcn_wmma_f32_16x16x32_f16(
            false, a, false, vb[ot], (short)0, oacc[ot], false, false);
    }
  }

  // epilogue: O * (1/l), store f16 into [B,S,D] (head-interleaved)
  float rinv[8];
#pragma unroll
  for (int i = 0; i < 8; ++i) rinv[i] = 1.0f / lrow[i];
  const int hi8 = (lane >> 4) * 8;
  const int nc  = lane & 15;
#pragma unroll
  for (int ot = 0; ot < 4; ++ot)
#pragma unroll
    for (int i = 0; i < 8; ++i) {
      const int s   = q0 + wave * 16 + i + hi8;
      const int col = hh * HD_ + ot * 16 + nc;
      const float val = oacc[ot][i] * rinv[i];
      Ob[((size_t)bb * S_ + s) * D_ + col] = (_Float16)val;
    }
}

extern "C" void kernel_launch(void* const* d_in, const int* in_sizes, int n_in,
                              void* d_out, int out_size, void* d_ws, size_t ws_size,
                              hipStream_t stream) {
  (void)in_sizes; (void)n_in; (void)out_size; (void)ws_size;
  const float* q  = (const float*)d_in[0];
  const float* k  = (const float*)d_in[1];
  const float* v  = (const float*)d_in[2];
  const float* Wq = (const float*)d_in[3];
  const float* bq = (const float*)d_in[4];
  const float* Wo = (const float*)d_in[5];
  const float* bo = (const float*)d_in[6];
  float* out = (float*)d_out;

  const size_t seg = (size_t)B_ * H_ * S_ * HD_;  // 4,194,304 f16 elems = 8 MB
  _Float16* Qh = (_Float16*)d_ws;
  _Float16* Kh = Qh + seg;
  _Float16* Vt = Kh + seg;
  _Float16* Ob = Vt + seg;

  const dim3 blk(128);
  const dim3 gproj(D_ / 64, (B_ * S_) / 128);
  // bug-faithful: all three use Wq/bq
  gemm512<true, 0><<<gproj, blk, 0, stream>>>((const void*)q, Wq, bq, Qh, nullptr);
  gemm512<true, 0><<<gproj, blk, 0, stream>>>((const void*)k, Wq, bq, Kh, nullptr);
  gemm512<true, 1><<<gproj, blk, 0, stream>>>((const void*)v, Wq, bq, Vt, nullptr);

  flash_attn<<<dim3(S_ / 64, B_ * H_), blk, 0, stream>>>(Qh, Kh, Vt, Ob);

  gemm512<false, 2><<<gproj, blk, 0, stream>>>((const void*)Ob, Wo, bo, nullptr, out);
}
